// PtrNet1_8229157339374
// MI455X (gfx1250) — matrix operand order repaired and checked by reading, not compiled, not run on hardware
//
#include <hip/hip_runtime.h>
#include <hip/hip_bf16.h>
#include <cstdint>

typedef __attribute__((ext_vector_type(16))) _Float16 v16h;
typedef __attribute__((ext_vector_type(8)))  _Float16 v8h;
typedef __attribute__((ext_vector_type(8)))  float    v8f;

#define B_  16
#define J_  20
#define M_  10
#define P_  8
#define H_  128
#define NH_ 2
#define EC_ 3
#define N_  202   // J*M + 2
#define NB_ 200   // J*M

// ---------------- WMMA helper (CDNA5 gfx1250, wave32) ----------------
__device__ __forceinline__ v8f wmma_f16(v16h a, v16h b, v8f c) {
  // v_wmma_f32_16x16x32_f16: D = A(16x32 f16) x B(32x16 f16) + C(16x16 f32)
  return __builtin_amdgcn_wmma_f32_16x16x32_f16(false, a, false, b, (short)0, c,
                                                false, false);
}

// LDS offset (raw 32-bit LDS address) of a __shared__ object, for async-DMA dest.
__device__ __forceinline__ uint32_t lds_addr_of(const void* p) {
  return (uint32_t)(uintptr_t)(const __attribute__((address_space(3))) void*)p;
}

// Fragment-order layouts (per 16x16 output tile, K-block granularity 32):
//  A-frag (16x32 f16): slot = kb*32 + lane ; lane = 16*lh + row,
//    element i -> k' = 16*lh + i
//  B-frag (32x16 f16): slot = (kb*nct + ct)*32 + lane ; lane = 16*lh + col,
//    element q -> k' = 16*lh + q
// Each lane's 16 halves are 32 contiguous, 32B-aligned bytes in LDS.

// ---------------- Generic staged WMMA GEMM ----------------
// C[batch] (M x 128) = act( A[batch](M x K) @ B[batch](K x 128) [+ C] )
// A rows contiguous with stride K. B optionally transposed (Bsrc[j*K+kk]).
__global__ __launch_bounds__(256) void k_gemm(
    const float* __restrict__ A, const float* __restrict__ Bsrc,
    float* __restrict__ C, int M, int K, int KP, int nRowTiles,
    long strideA, int aDiv, long strideB, int bMod, long strideC,
    int btrans, int accum, int relu) {
  __shared__ alignas(32) _Float16 sBm[224 * 128];  // frag order, 8 col-tiles
  __shared__ alignas(32) _Float16 sAm[16 * 224];   // frag order
  int tid = threadIdx.x;
  int batch = blockIdx.x / nRowTiles;
  int rt = blockIdx.x % nRowTiles;
  const float* Ap = A + (long)(batch / aDiv) * strideA;
  const float* Bp = Bsrc + (long)(batch % bMod) * strideB;
  float* Cp = C + (long)batch * strideC;
  int KB = KP >> 5;

  // stage B in fragment order: one v16h (one lane-fragment) per iteration
  for (int f = tid; f < KB * 256; f += 256) {
    int kb = f >> 8;
    int rem = f & 255;           // ct*32 + lane
    int lane = rem & 31, ct = rem >> 5;
    int lh = lane >> 4, c = lane & 15;
    int j = ct * 16 + c;
    int kk0 = kb * 32 + 16 * lh;
    v16h tmp;
#pragma unroll
    for (int q = 0; q < 16; ++q) {
      int kk = kk0 + q;
      float v = 0.f;
      if (kk < K) v = btrans ? Bp[(long)j * K + kk] : Bp[(long)kk * 128 + j];
      tmp[q] = (_Float16)v;
    }
    *(v16h*)(sBm + (size_t)f * 16) = tmp;
  }
  // stage A in fragment order: 8-half segments
  int r0 = rt * 16;
  for (int sgi = tid; sgi < 2 * KP; sgi += 256) {
    int pos = sgi * 8;           // linear (r,kk), row length KP
    int r = pos / KP;
    int kk0 = pos - r * KP;      // multiple of 8
    int kb = kk0 >> 5, k0 = kk0 & 31;
    int seg = k0 >> 3;
    int lh = seg >> 1, ip = (seg & 1) * 8;
    v8h tmp;
#pragma unroll
    for (int i2 = 0; i2 < 8; ++i2) {
      int kk = kk0 + i2;
      float v = 0.f;
      if (r0 + r < M && kk < K) v = Ap[(long)(r0 + r) * K + kk];
      tmp[i2] = (_Float16)v;
    }
    *(v8h*)(sAm + ((size_t)(kb * 32 + 16 * lh + r)) * 16 + ip) = tmp;
  }
  __syncthreads();

  int w = tid >> 5, lane = tid & 31;
  v8f acc = {};
#pragma unroll 4
  for (int kb = 0; kb < KB; ++kb) {
    v16h af = *(const v16h*)(sAm + (size_t)(kb * 32 + lane) * 16);
    v16h bf = *(const v16h*)(sBm + (size_t)((kb * 8 + w) * 32 + lane) * 16);
    acc = wmma_f16(af, bf, acc);
  }
  int lh = lane >> 4, col = w * 16 + (lane & 15);
#pragma unroll
  for (int v = 0; v < 8; ++v) {
    int row = r0 + v + 8 * lh;
    if (row < M) {
      long o = (long)row * 128 + col;
      float val = acc[v];
      if (accum) val += Cp[o];
      if (relu) val = fmaxf(val, 0.f);
      Cp[o] = val;
    }
  }
}

// ---------------- one-time: convert Vec (NH,128,128 f32) to f16 in fragment
// order per (m, colgrp): dst[((((m*4+cg)*4+kb)*2+ct)*32+lane)*16+q]
__global__ void k_prep_vec(const float* __restrict__ Vec,
                           _Float16* __restrict__ vf) {
  int idx = blockIdx.x * blockDim.x + threadIdx.x;
  if (idx >= NH_ * 128 * 128) return;
  int m = idx >> 14;
  int rem = idx & 16383;
  int h = rem >> 7, j = rem & 127;
  int cg = j >> 5, jl = j & 31, ct = jl >> 4, c = jl & 15;
  int kb = h >> 5, kp = h & 31, lh = kp >> 4, q = kp & 15;
  int lane = 16 * lh + c;
  vf[((((size_t)(m * 4 + cg) * 4 + kb) * 2 + ct) * 32 + lane) * 16 + q] =
      (_Float16)Vec[idx];
}

// ---------------- Fused glimpse ----------------
// e=tanh(q+u2); u=e@Vec (WMMA); masked softmax over n (per column); g=sum a*u2
// grid: (B*NH*4) blocks; block = one (b,m) pair and a 32-column group.
__global__ __launch_bounds__(256) void k_glimpse(
    const float* __restrict__ q, const float* __restrict__ u2,
    const _Float16* __restrict__ vec16, const float* __restrict__ mask,
    float* __restrict__ g) {
  __shared__ alignas(32) _Float16 sVec[4096];  // frag order [(kb*2+ct)*32+lane]*16
  __shared__ alignas(32) _Float16 sE[8192];    // frag order [(nt*4+kb)*32+lane]*16
  __shared__ float sU[208 * 32];               // [n][c]
  __shared__ float sQ[128];

  int tid = threadIdx.x;
  int colgrp = blockIdx.x & 3;
  int bm = blockIdx.x >> 2;
  int b = bm / NH_, m = bm - b * NH_;
  int colbase = colgrp * 32;

  // Async DMA the pre-swizzled Vec panel (8KB) straight into LDS (ASYNCcnt path)
  {
    const _Float16* vsrc = vec16 + (size_t)(m * 4 + colgrp) * 4096;
    uint32_t lbase = lds_addr_of(sVec);
    const _Float16* g0 = vsrc + tid * 8;
    uint32_t l0 = lbase + tid * 16;
    asm volatile("global_load_async_to_lds_b128 %0, %1, off"
                 :: "v"(l0), "v"(g0) : "memory");
    const _Float16* g1 = vsrc + 2048 + tid * 8;
    uint32_t l1 = lbase + 4096 + tid * 16;
    asm volatile("global_load_async_to_lds_b128 %0, %1, off"
                 :: "v"(l1), "v"(g1) : "memory");
  }
  if (tid < 128) sQ[tid] = q[bm * 128 + tid];
  asm volatile("s_wait_asynccnt 0x0" ::: "memory");
  __syncthreads();

  const float* u2bm = u2 + (long)bm * NB_ * 128;

  for (int s = 0; s < 4; ++s) {
    {  // stage e tile (64 rows x 128) as f16 fragments; thread = one (n', kb)
      int kb = tid & 3;
      int nloc = tid >> 2;
      int n = s * 64 + nloc;
      int r = nloc & 15, nt = nloc >> 4;
#pragma unroll
      for (int seg = 0; seg < 4; ++seg) {
        v8h tmp;
#pragma unroll
        for (int i2 = 0; i2 < 8; ++i2) {
          int h = kb * 32 + seg * 8 + i2;
          float v = 0.f;
          if (n < NB_) v = tanhf(sQ[h] + u2bm[(long)n * 128 + h]);
          tmp[i2] = (_Float16)v;
        }
        int lh = seg >> 1, ip = (seg & 1) * 8;
        *(v8h*)(sE + ((size_t)((nt * 4 + kb) * 32 + 16 * lh + r)) * 16 + ip) = tmp;
      }
      if (n + 64 < NB_)
        __builtin_prefetch(&u2bm[(long)(n + 64) * 128 + kb * 32], 0, 0);
    }
    __syncthreads();
    int w = tid >> 5, lane = tid & 31;
    int nt2 = w >> 1, ct = w & 1;
    int rowbase = s * 64 + nt2 * 16;
    if (rowbase < 208) {
      v8f acc = {};
#pragma unroll
      for (int kb2 = 0; kb2 < 4; ++kb2) {
        v16h af = *(const v16h*)(sE + (size_t)((nt2 * 4 + kb2) * 32 + lane) * 16);
        v16h bf = *(const v16h*)(sVec + (size_t)((kb2 * 2 + ct) * 32 + lane) * 16);
        acc = wmma_f16(af, bf, acc);
      }
      int lh = lane >> 4, c = ct * 16 + (lane & 15);
#pragma unroll
      for (int v = 0; v < 8; ++v)
        sU[(rowbase + v + 8 * lh) * 32 + c] = acc[v];
    }
    __syncthreads();
  }

  // per-column masked softmax over n, then weighted sum of u2
  if (tid < 32) {
    int c = tid;
    float mx = -3.0e38f;
    for (int n = 0; n < NB_; ++n) {
      float uv = (mask[b * NB_ + n] > 0.f) ? sU[n * 32 + c] : -1.0e9f;
      mx = fmaxf(mx, uv);
    }
    float sum = 0.f;
    for (int n = 0; n < NB_; ++n) {
      float uv = (mask[b * NB_ + n] > 0.f) ? sU[n * 32 + c] : -1.0e9f;
      sum += expf(uv - mx);
    }
    float ga = 0.f;
    for (int n = 0; n < NB_; ++n) {
      float uv = (mask[b * NB_ + n] > 0.f) ? sU[n * 32 + c] : -1.0e9f;
      ga += (expf(uv - mx) / sum) * u2bm[(long)n * 128 + colbase + c];
    }
    g[bm * 128 + colbase + c] = ga;
  }
}

// ---------------- small VALU kernels ----------------
__global__ void k_norm_adj(const float* __restrict__ adj, float* __restrict__ an) {
  int row = blockIdx.x * blockDim.x + threadIdx.x;
  if (row >= B_ * EC_ * N_) return;
  const float* src = adj + (long)row * N_;
  float* dst = an + (long)row * N_;
  float s = 0.f;
  for (int j = 0; j < N_; ++j) s += src[j];
  float inv = 1.f / fmaxf(s, 1e-8f);
  for (int j = 0; j < N_; ++j) dst[j] = src[j] * inv;
}

__global__ void k_hbar(const float* __restrict__ enc, float* __restrict__ hbar) {
  int i = blockIdx.x * blockDim.x + threadIdx.x;
  if (i >= B_ * H_) return;
  int b = i >> 7, k = i & 127;
  float s = 0.f;
  for (int n = 0; n < N_; ++n) s += enc[((long)b * N_ + n) * H_ + k];
  hbar[i] = s / (float)N_;
}

__global__ void k_init(const float* __restrict__ dec_input,
                       const float* __restrict__ v_f, float* dec_in, float* first,
                       int* count, float* out) {
  int i = blockIdx.x * blockDim.x + threadIdx.x;
  if (i < B_ * H_) {
    dec_in[i] = dec_input[i & 127];
    first[i] = v_f[i & 127];
  }
  if (i < B_ * J_) count[i] = 0;
  if (i < B_) out[i] = 0.f;
}

__global__ void k_mask(const int* __restrict__ count, float* __restrict__ mask) {
  int i = blockIdx.x * blockDim.x + threadIdx.x;
  if (i >= B_ * NB_) return;
  int b = i / NB_, n = i - b * NB_;
  int j = n / M_, pm = n - j * M_;
  mask[i] = (count[b * J_ + j] == pm) ? 1.f : 0.f;
}

__global__ void k_query1(const float* __restrict__ hbar,
                         const float* __restrict__ dec_in,
                         const float* __restrict__ first,
                         const float* __restrict__ Wq1, float* __restrict__ q) {
  int i = blockIdx.x * blockDim.x + threadIdx.x;
  if (i >= B_ * NH_ * H_) return;
  int b = i >> 8, r = i & 255;
  int m = r >> 7, hh = r & 127;
  const float* wrow = Wq1 + (long)(m * H_ + hh) * (3 * H_);
  float acc = 0.f;
  for (int c = 0; c < H_; ++c) acc += hbar[b * H_ + c] * wrow[c];
  for (int c = 0; c < H_; ++c) acc += dec_in[b * H_ + c] * wrow[H_ + c];
  for (int c = 0; c < H_; ++c) acc += first[b * H_ + c] * wrow[2 * H_ + c];
  q[i] = acc;
}

__global__ void k_mhe(const float* __restrict__ g, const float* __restrict__ mhe,
                      float* __restrict__ g2) {
  int i = blockIdx.x * blockDim.x + threadIdx.x;
  if (i >= B_ * H_) return;
  int b = i >> 7, k = i & 127;
  float acc = 0.f;
  for (int c = 0; c < NH_ * H_; ++c)
    acc += g[b * NH_ * H_ + c] * mhe[k * NH_ * H_ + c];
  g2[i] = acc;
}

__global__ void k_qnext(const float* __restrict__ g2, const float* __restrict__ W,
                        float* __restrict__ q) {
  int i = blockIdx.x * blockDim.x + threadIdx.x;
  if (i >= B_ * NH_ * H_) return;
  int b = i >> 8, r = i & 255;
  int m = r >> 7, k = r & 127;
  const float* wrow = W + (long)(m * H_ + k) * H_;
  float acc = 0.f;
  for (int h = 0; h < H_; ++h) acc += g2[b * H_ + h] * wrow[h];
  q[i] = acc;
}

__global__ void k_qp(const float* __restrict__ g2, const float* __restrict__ Wq2,
                     float* __restrict__ qp) {
  int i = blockIdx.x * blockDim.x + threadIdx.x;
  if (i >= B_ * H_) return;
  int b = i >> 7, k = i & 127;
  float acc = 0.f;
  for (int h = 0; h < H_; ++h) acc += g2[b * H_ + h] * Wq2[k * H_ + h];
  qp[i] = acc;
}

__global__ void k_pointer(const float* __restrict__ qp,
                          const float* __restrict__ u2p,
                          const float* __restrict__ Vec2,
                          const float* __restrict__ mask,
                          const float* __restrict__ enc, int* count,
                          float* dec_in, float* first, float* out, int t) {
  __shared__ float su[NB_];
  __shared__ int sidx;
  int b = blockIdx.x, tid = threadIdx.x;
  if (tid < NB_) {
    float acc = 0.f;
    const float* up = u2p + ((long)b * NB_ + tid) * H_;
    for (int h = 0; h < H_; ++h)
      acc += Vec2[h] * tanhf(qp[b * H_ + h] + up[h]);
    float u = 10.f * tanhf(acc);  // C_CLIP * tanh(u / TEMP), TEMP=1
    su[tid] = (mask[b * NB_ + tid] > 0.f) ? u : -1.0e9f;
  }
  __syncthreads();
  if (tid == 0) {
    float mx = -3.0e38f;
    int idx = 0;
    for (int n = 0; n < NB_; ++n)
      if (su[n] > mx) { mx = su[n]; idx = n; }
    float sum = 0.f;
    for (int n = 0; n < NB_; ++n) sum += expf(su[n] - mx);
    float lp = su[idx] - mx - logf(sum);
    out[b] += lp;
    out[B_ + b * NB_ + t] = (float)idx;
    count[b * J_ + idx / M_] += 1;
    sidx = idx;
  }
  __syncthreads();
  if (tid < H_) {
    float v = enc[((long)b * N_ + sidx) * H_ + tid];
    dec_in[b * H_ + tid] = v;
    if (t == 0) first[b * H_ + tid] = v;
  }
}

// ---------------- host orchestration ----------------
extern "C" void kernel_launch(void* const* d_in, const int* in_sizes, int n_in,
                              void* d_out, int out_size, void* d_ws,
                              size_t ws_size, hipStream_t stream) {
  (void)in_sizes; (void)n_in; (void)out_size; (void)ws_size;
  const float* nf    = (const float*)d_in[0];
  const float* adj   = (const float*)d_in[1];
  const float* W_emb = (const float*)d_in[2];
  const float* Wr1   = (const float*)d_in[3];
  const float* Ws1   = (const float*)d_in[4];
  const float* Wr2   = (const float*)d_in[5];
  const float* Ws2   = (const float*)d_in[6];
  const float* Wq1   = (const float*)d_in[7];
  const float* Wref1 = (const float*)d_in[8];
  const float* Vec1  = (const float*)d_in[9];
  const float* Wq3   = (const float*)d_in[10];
  const float* Wref3 = (const float*)d_in[11];
  const float* Vec3  = (const float*)d_in[12];
  const float* Wq4   = (const float*)d_in[13];
  const float* Wref4 = (const float*)d_in[14];
  const float* Vec4  = (const float*)d_in[15];
  const float* mhe   = (const float*)d_in[16];
  const float* Wq2   = (const float*)d_in[17];
  const float* Wref2 = (const float*)d_in[18];
  const float* Vec2  = (const float*)d_in[19];
  const float* decI  = (const float*)d_in[20];
  const float* v_f   = (const float*)d_in[21];
  float* out = (float*)d_out;

  char* ws = (char*)d_ws;
  size_t off = 0;
  auto alloc = [&](size_t nfl) {
    float* p = (float*)(ws + off);
    off += ((nfl * 4 + 255) / 256) * 256;
    return p;
  };
  float* an    = alloc((size_t)B_ * EC_ * N_ * N_);
  float* hA    = alloc((size_t)B_ * N_ * H_);
  float* hB    = alloc((size_t)B_ * N_ * H_);
  float* hw    = alloc((size_t)B_ * N_ * H_);
  float* hbar  = alloc(B_ * H_);
  float* u2_1  = alloc((size_t)B_ * NH_ * NB_ * H_);
  float* u2_3  = alloc((size_t)B_ * NH_ * NB_ * H_);
  float* u2_4  = alloc((size_t)B_ * NH_ * NB_ * H_);
  float* u2p   = alloc((size_t)B_ * NB_ * H_);
  float* qbuf  = alloc(B_ * NH_ * H_);
  float* gbuf  = alloc(B_ * NH_ * H_);
  float* g2    = alloc(B_ * H_);
  float* qpb   = alloc(B_ * H_);
  float* maskb = alloc(B_ * NB_);
  int* count   = (int*)alloc(B_ * J_);
  float* decb  = alloc(B_ * H_);
  float* firstb= alloc(B_ * H_);
  _Float16* vf1 = (_Float16*)alloc((size_t)NH_ * H_ * H_ / 2);
  _Float16* vf3 = (_Float16*)alloc((size_t)NH_ * H_ * H_ / 2);
  _Float16* vf4 = (_Float16*)alloc((size_t)NH_ * H_ * H_ / 2);

  auto gemm = [&](const float* A, const float* Bs, float* C, int M, int K,
                  int nbatch, long sA, int aDiv, long sB, int bMod, long sC,
                  int btr, int acc, int relu) {
    int KP = (K + 31) & ~31;
    int nrt = (M + 15) / 16;
    k_gemm<<<dim3(nbatch * nrt), dim3(256), 0, stream>>>(
        A, Bs, C, M, K, KP, nrt, sA, aDiv, sB, bMod, sC, btr, acc, relu);
  };

  // h0 = nf @ W_emb^T   (K=8 padded to 32)
  gemm(nf, W_emb, hA, B_ * N_, P_, 1, 0, 1, 0, 1, 0, /*btr*/1, 0, 0);
  k_norm_adj<<<dim3((B_ * EC_ * N_ + 255) / 256), dim3(256), 0, stream>>>(adj, an);

  // two GCRN layers
  const float* WrL[2] = {Wr1, Wr2};
  const float* WsL[2] = {Ws1, Ws2};
  float* src = hA;
  float* dst = hB;
  for (int L = 0; L < 2; ++L) {
    gemm(src, WsL[L], dst, B_ * N_, H_, 1, 0, 1, 0, 1, 0, 0, 0, 0);  // h@Ws
    for (int e = 0; e < EC_; ++e) {
      gemm(src, WrL[L] + (long)e * H_ * H_, hw, B_ * N_, H_, 1, 0, 1, 0, 1, 0,
           0, 0, 0);  // hw = h @ Wr[e]
      // msg += an_e @ hw ; ReLU fused on last edge class
      gemm(an + (long)e * N_ * N_, hw, dst, N_, N_, B_, (long)EC_ * N_ * N_, 1,
           (long)N_ * H_, B_, (long)N_ * H_, 0, 1, (e == EC_ - 1) ? 1 : 0);
    }
    float* tmp = src; src = dst; dst = tmp;
  }
  float* enc = src;  // == hA after two swaps
  k_hbar<<<dim3(8), dim3(256), 0, stream>>>(enc, hbar);

  // u2_X[b,m] = ref[b] @ WrefX[m]^T   (batched over b*NH, ref rows 0..199)
  gemm(enc, Wref1, u2_1, NB_, H_, B_ * NH_, (long)N_ * H_, NH_, (long)H_ * H_,
       NH_, (long)NB_ * H_, 1, 0, 0);
  gemm(enc, Wref3, u2_3, NB_, H_, B_ * NH_, (long)N_ * H_, NH_, (long)H_ * H_,
       NH_, (long)NB_ * H_, 1, 0, 0);
  gemm(enc, Wref4, u2_4, NB_, H_, B_ * NH_, (long)N_ * H_, NH_, (long)H_ * H_,
       NH_, (long)NB_ * H_, 1, 0, 0);
  gemm(enc, Wref2, u2p, NB_, H_, B_, (long)N_ * H_, 1, 0, 1, (long)NB_ * H_, 1,
       0, 0);

  // pre-swizzle Vec tensors into WMMA fragment order (f16) for async staging
  k_prep_vec<<<dim3(128), dim3(256), 0, stream>>>(Vec1, vf1);
  k_prep_vec<<<dim3(128), dim3(256), 0, stream>>>(Vec3, vf3);
  k_prep_vec<<<dim3(128), dim3(256), 0, stream>>>(Vec4, vf4);

  k_init<<<dim3(8), dim3(256), 0, stream>>>(decI, v_f, decb, firstb, count, out);

  for (int t = 0; t < NB_; ++t) {
    k_mask<<<dim3(13), dim3(256), 0, stream>>>(count, maskb);
    k_query1<<<dim3(16), dim3(256), 0, stream>>>(hbar, decb, firstb, Wq1, qbuf);
    k_glimpse<<<dim3(B_ * NH_ * 4), dim3(256), 0, stream>>>(qbuf, u2_1, vf1,
                                                            maskb, gbuf);
    k_mhe<<<dim3(8), dim3(256), 0, stream>>>(gbuf, mhe, g2);
    k_qnext<<<dim3(16), dim3(256), 0, stream>>>(g2, Wq3, qbuf);
    k_glimpse<<<dim3(B_ * NH_ * 4), dim3(256), 0, stream>>>(qbuf, u2_3, vf3,
                                                            maskb, gbuf);
    k_mhe<<<dim3(8), dim3(256), 0, stream>>>(gbuf, mhe, g2);
    k_qnext<<<dim3(16), dim3(256), 0, stream>>>(g2, Wq4, qbuf);
    k_glimpse<<<dim3(B_ * NH_ * 4), dim3(256), 0, stream>>>(qbuf, u2_4, vf4,
                                                            maskb, gbuf);
    k_mhe<<<dim3(8), dim3(256), 0, stream>>>(gbuf, mhe, g2);
    k_qp<<<dim3(8), dim3(256), 0, stream>>>(g2, Wq2, qpb);
    k_pointer<<<dim3(16), dim3(256), 0, stream>>>(qpb, u2p, Vec2, maskb, enc,
                                                  count, decb, firstb, out, t);
  }
}
